// MemoryExpert_7438883357036
// MI455X (gfx1250) — matrix operand order repaired and checked by reading, not compiled, not run on hardware
//
#include <hip/hip_runtime.h>

// ---- problem constants (match reference) ----
#define H_  1024
#define M_  1024
#define NH_ 16
#define DH_ 64
#define CR_ 4
#define B_  4
#define S_  2048
#define L_  (S_/CR_)

typedef __bf16 bf16;
typedef __attribute__((ext_vector_type(16))) __bf16        v16bf;
typedef __attribute__((ext_vector_type(8)))  float         v8f;
typedef __attribute__((ext_vector_type(4)))  unsigned int  u32x4;
typedef __attribute__((ext_vector_type(8)))  int           i32x8;
typedef __attribute__((ext_vector_type(4)))  int           i32x4;

#if defined(__gfx1250__)
#define ON_GFX1250 1
#else
#define ON_GFX1250 0
#endif

#define NEG_INF (-3.402823466e38f)

__device__ __forceinline__ float sigmoidf_(float x) { return 1.f / (1.f + __expf(-x)); }

// =====================================================================
// Tensor Data Mover: 2D bf16 tile (tile_x cols × tile_y rows) -> LDS
// D# group layouts per CDNA5 ISA ch.8 (data_size code 1 = 2 bytes)
// clang-23 toolchain: 6-arg builtin
// =====================================================================
#if ON_GFX1250 && __has_builtin(__builtin_amdgcn_tensor_load_to_lds)
#define HAVE_TDM 1
__device__ __forceinline__ void tdm_load_tile_bf16(
    unsigned lds_byte_addr, const bf16* tile_start,
    unsigned tile_x, unsigned tile_y,
    unsigned long long tensor_d0, unsigned long long tensor_d1,
    unsigned long long row_stride_elems)
{
  unsigned long long ga = (unsigned long long)(const void*)tile_start;
  u32x4 g0;
  g0[0] = 1u;                                            // count=1, user mode
  g0[1] = lds_byte_addr;                                 // lds_addr
  g0[2] = (unsigned)(ga & 0xFFFFFFFFu);                  // global_addr[31:0]
  g0[3] = (unsigned)((ga >> 32) & 0x01FFFFFFu) | (2u << 30); // addr[56:32] | type=2
  i32x8 g1;
  g1[0] = (int)(1u << 16);                               // data_size=1 (2B)
  g1[1] = (int)((tensor_d0 & 0xFFFFu) << 16);            // tensor_dim0[15:0]
  g1[2] = (int)(((tensor_d0 >> 16) & 0xFFFFu) | ((tensor_d1 & 0xFFFFu) << 16));
  g1[3] = (int)(((tensor_d1 >> 16) & 0xFFFFu) | ((tile_x & 0xFFFFu) << 16)); // tile_dim0
  g1[4] = (int)(tile_y & 0xFFFFu);                       // tile_dim1 (tile_dim2=0)
  g1[5] = (int)(row_stride_elems & 0xFFFFFFFFu);         // tensor_dim0_stride lo
  g1[6] = (int)((row_stride_elems >> 32) & 0xFFFFu);     // stride hi (dim1_stride=0)
  g1[7] = 0;
  i32x4 z4; z4[0]=0; z4[1]=0; z4[2]=0; z4[3]=0;
  i32x8 z8; z8[0]=0; z8[1]=0; z8[2]=0; z8[3]=0; z8[4]=0; z8[5]=0; z8[6]=0; z8[7]=0;
  __builtin_amdgcn_tensor_load_to_lds(g0, g1, z4, z4, z8, 0);
}
#else
#define HAVE_TDM 0
#endif

// =====================================================================
// elementwise converts
// =====================================================================
__global__ void cvt_f32_bf16_kernel(const float* __restrict__ in, bf16* __restrict__ out, int n) {
  int i = blockIdx.x * 256 + threadIdx.x;
  if (i < n) out[i] = (bf16)in[i];
}

// W[K][N] f32 -> Wt[N][K] bf16
__global__ void wtrans_bf16_kernel(const float* __restrict__ W, bf16* __restrict__ Wt, int K, int N) {
  int i = blockIdx.x * 256 + threadIdx.x;
  if (i < K * N) {
    int k = i / N, n = i % N;
    Wt[(size_t)n * K + k] = (bf16)W[i];
  }
}

// =====================================================================
// WMMA GEMM: C[M][N] = act(A[M][K] @ Wt[N][K]^T + bias)
// block tile 128x128x32, 8 waves (4 along M, 2 along N), 2x4 frags/wave
// Double-buffered LDS, TDM prefetch pipelined one k-tile ahead:
//   issue tile k+1, then s_wait_tensorcnt<=2 (current pair landed).
// A can be a concat of A0 (k < Ksplit, row stride ld0) and A1 (row stride ld1)
// =====================================================================
__launch_bounds__(256)
__global__ void gemm_wmma_bf16_kernel(
    const bf16* __restrict__ A0, int ld0, int Ksplit,
    const bf16* __restrict__ A1, int ld1,
    const bf16* __restrict__ Wt,                // [N][K] row-major, stride Ktot
    const float* __restrict__ bias,
    float* __restrict__ C32, bf16* __restrict__ Cbf,
    int Mtot, int Ntot, int Ktot, int act)
{
  __shared__ __align__(128) bf16 As[2][128 * 32];
  __shared__ __align__(128) bf16 Bs[2][128 * 32];

  const int tid  = threadIdx.x;
  const int lane = tid & 31;
  const int wid  = tid >> 5;
  const int r    = lane & 15;
  const int half = lane >> 4;
  const int wm   = wid & 3;     // 0..3 -> 32-row slabs
  const int wn   = wid >> 2;    // 0..1 -> 64-col slabs
  const int mb   = blockIdx.y * 128;
  const int nb   = blockIdx.x * 128;
  const int nk   = Ktot / 32;

  v8f acc[2][4];
#pragma unroll
  for (int mt = 0; mt < 2; mt++)
#pragma unroll
    for (int nt = 0; nt < 4; nt++)
#pragma unroll
      for (int j = 0; j < 8; j++) acc[mt][nt][j] = 0.f;

#if HAVE_TDM
  auto issue_tiles = [&](int kbx, int bufi) {
    const bf16* Ab; int ka, lda;
    if (kbx < Ksplit) { Ab = A0; ka = kbx;          lda = ld0; }
    else              { Ab = A1; ka = kbx - Ksplit; lda = ld1; }
    tdm_load_tile_bf16((unsigned)(unsigned long long)(const void*)&As[bufi][0],
                       Ab + (size_t)mb * lda + ka, 32u, 128u,
                       (unsigned long long)lda, (unsigned long long)Mtot,
                       (unsigned long long)lda);
    tdm_load_tile_bf16((unsigned)(unsigned long long)(const void*)&Bs[bufi][0],
                       Wt + (size_t)nb * Ktot + kbx, 32u, 128u,
                       (unsigned long long)Ktot, (unsigned long long)Ntot,
                       (unsigned long long)Ktot);
  };
  if (wid == 0) issue_tiles(0, 0);
#endif

  for (int kt = 0; kt < nk; kt++) {
    const int cur = kt & 1;
#if HAVE_TDM
    if (wid == 0) {
      if (kt + 1 < nk) {
        issue_tiles((kt + 1) * 32, cur ^ 1);
        __builtin_amdgcn_s_wait_tensorcnt(2);   // current pair complete, next in flight
      } else {
        __builtin_amdgcn_s_wait_tensorcnt(0);
      }
    }
#else
    {
      int kb = kt * 32;
      const bf16* Ab; int ka, lda;
      if (kb < Ksplit) { Ab = A0; ka = kb;          lda = ld0; }
      else             { Ab = A1; ka = kb - Ksplit; lda = ld1; }
      for (int idx = tid; idx < 128 * 32; idx += 256) {
        int row = idx >> 5, k = idx & 31;
        As[cur][idx] = Ab[(size_t)(mb + row) * lda + ka + k];
        Bs[cur][idx] = Wt[(size_t)(nb + row) * Ktot + kb + k];
      }
    }
#endif
    __syncthreads();

    // A fragment: lane=row r, elems [0..15]=K(half*16 .. half*16+15)
    v16bf af[2];
#pragma unroll
    for (int mt = 0; mt < 2; mt++) {
      const bf16* ap = &As[cur][(wm * 32 + mt * 16 + r) * 32];
#pragma unroll
      for (int i = 0; i < 8; i++) { af[mt][i] = ap[half * 16 + i]; af[mt][8 + i] = ap[half * 16 + 8 + i]; }
    }
    // B fragment: lane=col r, elems [0..15]=K(half*16 .. half*16+15)
    v16bf bfr[4];
#pragma unroll
    for (int nt = 0; nt < 4; nt++) {
      const bf16* bp = &Bs[cur][(wn * 64 + nt * 16 + r) * 32];
#pragma unroll
      for (int j = 0; j < 16; j++) bfr[nt][j] = bp[half * 16 + j];
    }
#pragma unroll
    for (int mt = 0; mt < 2; mt++)
#pragma unroll
      for (int nt = 0; nt < 4; nt++)
        acc[mt][nt] = __builtin_amdgcn_wmma_f32_16x16x32_bf16(
            false, af[mt], false, bfr[nt], (short)0, acc[mt][nt], false, false);
    __syncthreads();   // all reads of buf[cur] retired before it is re-filled
  }

  // epilogue: acc elem j on lane -> m = j + 8*half, n = r  (within 16x16 tile)
#pragma unroll
  for (int mt = 0; mt < 2; mt++)
#pragma unroll
    for (int nt = 0; nt < 4; nt++)
#pragma unroll
      for (int j = 0; j < 8; j++) {
        int row = mb + wm * 32 + mt * 16 + j + 8 * half;
        int col = nb + wn * 64 + nt * 16 + r;
        float v = acc[mt][nt][j] + (bias ? bias[col] : 0.f);
        if (act == 1) v = v * sigmoidf_(v);   // SiLU
        if (C32) C32[(size_t)row * Ntot + col] = v;
        if (Cbf) Cbf[(size_t)row * Ntot + col] = (bf16)v;
      }
}

// =====================================================================
// flash attention: one wave per 16-query tile, key tiles of 32, DH=64
// All 4 waves of a block share (b,h) -> K/V tiles staged once per block
// in LDS via TDM (row stride NH*DH). q/k/v/out layout: [B,S|M,NH,DH] bf16
// =====================================================================
__launch_bounds__(128)
__global__ void attn_wmma_kernel(const bf16* __restrict__ Q, const bf16* __restrict__ Kc,
                                 const bf16* __restrict__ Vc,
                                 const unsigned char* __restrict__ mask,
                                 bf16* __restrict__ Oa)
{
  __shared__ __align__(128) bf16 Ks[32 * 64];
  __shared__ __align__(128) bf16 Vs[32 * 64];
  __shared__ __align__(64)  bf16 Pst[4][16 * 32];
  const int tid = threadIdx.x, lane = tid & 31, wid = tid >> 5;
  const int r = lane & 15, half = lane >> 4;
  const int QT = S_ / 16;
  int qt    = blockIdx.x * 4 + wid;
  int qtile = qt % QT;
  int bh    = qt / QT;          // identical for all 4 waves in the block
  int h     = bh % NH_;
  int b     = bh / NH_;
  int qbase = qtile * 16;
  const float scale = 0.125f;   // DH^-0.5

  // Q fragments (two K=32 chunks of d), straight from global
  const bf16* qrow = Q + ((size_t)(b * S_ + qbase + r) * NH_ + h) * DH_;
  v16bf a0, a1;
#pragma unroll
  for (int i = 0; i < 8; i++) {
    a0[i] = qrow[half * 16 + i];       a0[8 + i] = qrow[half * 16 + 8 + i];
    a1[i] = qrow[32 + half * 16 + i];  a1[8 + i] = qrow[40 + half * 16 + i];
  }

  float mrun[8], lrun[8];
  v8f o[4];
#pragma unroll
  for (int j = 0; j < 8; j++) { mrun[j] = NEG_INF; lrun[j] = 0.f; }
#pragma unroll
  for (int dt = 0; dt < 4; dt++)
#pragma unroll
    for (int j = 0; j < 8; j++) o[dt][j] = 0.f;

  bf16* P = &Pst[wid][0];

  for (int kb = 0; kb < M_; kb += 32) {
#if HAVE_TDM
    if (wid == 0) {
      tdm_load_tile_bf16((unsigned)(unsigned long long)(const void*)Ks,
                         Kc + ((size_t)(b * M_ + kb) * NH_ + h) * DH_,
                         64u, 32u, 64ull, 32ull, (unsigned long long)(NH_ * DH_));
      tdm_load_tile_bf16((unsigned)(unsigned long long)(const void*)Vs,
                         Vc + ((size_t)(b * M_ + kb) * NH_ + h) * DH_,
                         64u, 32u, 64ull, 32ull, (unsigned long long)(NH_ * DH_));
      __builtin_amdgcn_s_wait_tensorcnt(0);
    }
#else
    for (int i = tid; i < 32 * 64; i += 128) {
      int key = i >> 6, d = i & 63;
      Ks[i] = Kc[((size_t)(b * M_ + kb + key) * NH_ + h) * DH_ + d];
      Vs[i] = Vc[((size_t)(b * M_ + kb + key) * NH_ + h) * DH_ + d];
    }
#endif
    __syncthreads();   // tiles visible to all waves

    float sc[2][8];
#pragma unroll
    for (int nh2 = 0; nh2 < 2; nh2++) {
      const bf16* kp = &Ks[(nh2 * 16 + r) * 64];
      v16bf b0, b1;
#pragma unroll
      for (int j = 0; j < 16; j++) { b0[j] = kp[half * 16 + j]; b1[j] = kp[32 + half * 16 + j]; }
      v8f s;
#pragma unroll
      for (int j = 0; j < 8; j++) s[j] = 0.f;
      s = __builtin_amdgcn_wmma_f32_16x16x32_bf16(false, a0, false, b0, (short)0, s, false, false);
      s = __builtin_amdgcn_wmma_f32_16x16x32_bf16(false, a1, false, b1, (short)0, s, false, false);
      bool mk = mask[(size_t)b * M_ + kb + nh2 * 16 + r] != 0;
#pragma unroll
      for (int j = 0; j < 8; j++) sc[nh2][j] = mk ? s[j] * scale : NEG_INF;
    }
    // online softmax; row m = j + 8*half lives on the 16 lanes of this half
#pragma unroll
    for (int j = 0; j < 8; j++) {
      float tmax = fmaxf(sc[0][j], sc[1][j]);
      for (int mk2 = 1; mk2 < 16; mk2 <<= 1) tmax = fmaxf(tmax, __shfl_xor(tmax, mk2, 16));
      float mnew  = fmaxf(mrun[j], tmax);
      float alpha = (mrun[j] > -1e30f) ? __expf(mrun[j] - mnew) : 0.f;
      float p0 = (sc[0][j] > -1e30f) ? __expf(sc[0][j] - mnew) : 0.f;
      float p1 = (sc[1][j] > -1e30f) ? __expf(sc[1][j] - mnew) : 0.f;
      float rs = p0 + p1;
      for (int mk2 = 1; mk2 < 16; mk2 <<= 1) rs += __shfl_xor(rs, mk2, 16);
      lrun[j] = lrun[j] * alpha + rs;
      mrun[j] = mnew;
#pragma unroll
      for (int dt = 0; dt < 4; dt++) o[dt][j] *= alpha;
      int mrow = j + 8 * half;
      P[mrow * 32 + r]      = (bf16)p0;   // per-wave slab: same-wave LDS ops are in-order
      P[mrow * 32 + 16 + r] = (bf16)p1;
    }
    // reload P as an A fragment (16 rows x 32 keys)
    v16bf pa;
    {
      const bf16* pp = &P[r * 32];
#pragma unroll
      for (int i = 0; i < 8; i++) { pa[i] = pp[half * 16 + i]; pa[8 + i] = pp[half * 16 + 8 + i]; }
    }
#pragma unroll
    for (int dt = 0; dt < 4; dt++) {
      v16bf vb;
      const bf16* vp = &Vs[(half * 16) * 64 + dt * 16 + r];
#pragma unroll
      for (int j = 0; j < 16; j++) vb[j] = vp[j * 64];
      o[dt] = __builtin_amdgcn_wmma_f32_16x16x32_bf16(false, pa, false, vb, (short)0, o[dt], false, false);
    }
    __syncthreads();   // everyone done with Ks/Vs before next tile overwrites
  }

#pragma unroll
  for (int j = 0; j < 8; j++) {
    float inv = (lrun[j] > 0.f) ? 1.f / lrun[j] : 0.f;  // nan_to_num for fully-masked rows
    int q = qbase + j + 8 * half;
    bf16* op = Oa + ((size_t)(b * S_ + q) * NH_ + h) * DH_;
#pragma unroll
    for (int dt = 0; dt < 4; dt++) op[dt * 16 + r] = (bf16)(o[dt][j] * inv);
  }
}

// =====================================================================
// read gate L2 + residual + LayerNorm (one block per row)
// =====================================================================
__launch_bounds__(256)
__global__ void rg_ln_kernel(const float* __restrict__ hs, const float* __restrict__ ret,
                             const float* __restrict__ rg1, const float* __restrict__ Wrg2,
                             const float* __restrict__ brg2, const float* __restrict__ gamma,
                             const float* __restrict__ beta, float* __restrict__ out)
{
  __shared__ float red[256];
  const int row = blockIdx.x, tid = threadIdx.x;

  float acc = 0.f;
  for (int i = tid; i < H_ / 2; i += 256) acc += rg1[(size_t)row * (H_ / 2) + i] * Wrg2[i];
  red[tid] = acc; __syncthreads();
  for (int s = 128; s > 0; s >>= 1) { if (tid < s) red[tid] += red[tid + s]; __syncthreads(); }
  float rg = sigmoidf_(red[0] + brg2[0]);
  __syncthreads();

  float y[4], sum = 0.f, sq = 0.f;
#pragma unroll
  for (int t = 0; t < 4; t++) {
    int i = tid + t * 256;
    float v = hs[(size_t)row * H_ + i] + rg * ret[(size_t)row * H_ + i];
    y[t] = v; sum += v; sq += v * v;
  }
  red[tid] = sum; __syncthreads();
  for (int s = 128; s > 0; s >>= 1) { if (tid < s) red[tid] += red[tid + s]; __syncthreads(); }
  float mu = red[0] / H_; __syncthreads();
  red[tid] = sq; __syncthreads();
  for (int s = 128; s > 0; s >>= 1) { if (tid < s) red[tid] += red[tid + s]; __syncthreads(); }
  float var = red[0] / H_ - mu * mu;
  float rstd = rsqrtf(var + 1e-5f);
#pragma unroll
  for (int t = 0; t < 4; t++) {
    int i = tid + t * 256;
    out[(size_t)row * H_ + i] = (y[t] - mu) * rstd * gamma[i] + beta[i];
  }
}

// write gate L2 (one wave per row)
__launch_bounds__(256)
__global__ void wg_kernel(const float* __restrict__ wg1, const float* __restrict__ Wwg2,
                          const float* __restrict__ bwg2, float* __restrict__ wgv)
{
  const int tid = threadIdx.x, lane = tid & 31, wid = tid >> 5;
  int row = blockIdx.x * 8 + wid;
  float acc = 0.f;
  for (int i = lane; i < H_ / 2; i += 32) acc += wg1[(size_t)row * (H_ / 2) + i] * Wwg2[i];
  for (int m = 1; m < 32; m <<= 1) acc += __shfl_xor(acc, m, 32);
  if (lane == 0) wgv[row] = sigmoidf_(acc + bwg2[0]);
}

// circular-buffer scatter of gated compressed rows into new_bank
__global__ void scatter_kernel(const float* __restrict__ c2, const float* __restrict__ wgv,
                               const int* __restrict__ wp, float* __restrict__ bank_out)
{
  int bl = blockIdx.x;
  int b = bl / L_, l = bl % L_;
  float wgm = 0.25f * (wgv[b * S_ + l * 4] + wgv[b * S_ + l * 4 + 1] +
                       wgv[b * S_ + l * 4 + 2] + wgv[b * S_ + l * 4 + 3]);
  int idx = (wp[0] + l) % M_;
  for (int i = threadIdx.x; i < H_; i += 256)
    bank_out[((size_t)b * M_ + idx) * H_ + i] = c2[((size_t)b * L_ + l) * H_ + i] * wgm;
}

__global__ void maskptr_kernel(const unsigned char* __restrict__ mask_in, const int* __restrict__ wp,
                               float* __restrict__ mask_out, float* __restrict__ ptr_out)
{
  int i = blockIdx.x * 256 + threadIdx.x;
  if (i < B_ * M_) {
    int m = i % M_;
    int d = (m - (wp[0] % M_) + M_) % M_;
    mask_out[i] = (d < L_ || mask_in[i]) ? 1.f : 0.f;
  }
  if (i == 0) ptr_out[0] = (float)((wp[0] + L_) % M_);
}

// =====================================================================
// host
// =====================================================================
extern "C" void kernel_launch(void* const* d_in, const int* in_sizes, int n_in,
                              void* d_out, int out_size, void* d_ws, size_t ws_size,
                              hipStream_t stream) {
  (void)in_sizes; (void)n_in; (void)out_size; (void)ws_size;
  const float* hs   = (const float*)d_in[0];
  const float* bank = (const float*)d_in[1];
  const unsigned char* mask = (const unsigned char*)d_in[2];
  const int*   wp   = (const int*)d_in[3];
  const float* Wc1  = (const float*)d_in[4];  const float* bc1  = (const float*)d_in[5];
  const float* Wc2  = (const float*)d_in[6];  const float* bc2  = (const float*)d_in[7];
  const float* Wq   = (const float*)d_in[8];  const float* bq   = (const float*)d_in[9];
  const float* Wk   = (const float*)d_in[10]; const float* bk   = (const float*)d_in[11];
  const float* Wv   = (const float*)d_in[12]; const float* bv   = (const float*)d_in[13];
  const float* Wo   = (const float*)d_in[14]; const float* bo   = (const float*)d_in[15];
  const float* Wwg1 = (const float*)d_in[16]; const float* bwg1 = (const float*)d_in[17];
  const float* Wwg2 = (const float*)d_in[18]; const float* bwg2 = (const float*)d_in[19];
  const float* Wrg1 = (const float*)d_in[20]; const float* brg1 = (const float*)d_in[21];
  const float* Wrg2 = (const float*)d_in[22]; const float* brg2 = (const float*)d_in[23];
  const float* lng  = (const float*)d_in[24]; const float* lnb  = (const float*)d_in[25];

  float* out_main = (float*)d_out;
  float* out_bank = out_main + (size_t)B_ * S_ * H_;
  float* out_mask = out_bank + (size_t)B_ * M_ * H_;
  float* out_ptr  = out_mask + (size_t)B_ * M_;

  // workspace carve-up
  char* wsp = (char*)d_ws; size_t off = 0;
  auto alloc = [&](size_t bytes) -> void* {
    off = (off + 255) & ~(size_t)255;
    void* p = wsp + off; off += bytes; return p;
  };
  bf16* hsb    = (bf16*)alloc((size_t)B_ * S_ * H_ * 2);
  bf16* bankb  = (bf16*)alloc((size_t)B_ * M_ * H_ * 2);
  bf16* qb     = (bf16*)alloc((size_t)B_ * S_ * H_ * 2);
  bf16* kb_    = (bf16*)alloc((size_t)B_ * M_ * H_ * 2);
  bf16* vb_    = (bf16*)alloc((size_t)B_ * M_ * H_ * 2);
  bf16* attb   = (bf16*)alloc((size_t)B_ * S_ * H_ * 2);
  bf16* retb   = (bf16*)alloc((size_t)B_ * S_ * H_ * 2);
  bf16* c1b    = (bf16*)alloc((size_t)B_ * L_ * H_ * 2);
  bf16* WqT    = (bf16*)alloc((size_t)H_ * H_ * 2);
  bf16* WkT    = (bf16*)alloc((size_t)H_ * H_ * 2);
  bf16* WvT    = (bf16*)alloc((size_t)H_ * H_ * 2);
  bf16* WoT    = (bf16*)alloc((size_t)H_ * H_ * 2);
  bf16* Wrg1T  = (bf16*)alloc((size_t)2 * H_ * (H_ / 2) * 2);
  bf16* Wwg1T  = (bf16*)alloc((size_t)H_ * (H_ / 2) * 2);
  bf16* Wc1T   = (bf16*)alloc((size_t)CR_ * H_ * H_ * 2);
  bf16* Wc2T   = (bf16*)alloc((size_t)H_ * H_ * 2);
  float* retf  = (float*)alloc((size_t)B_ * S_ * H_ * 4);
  float* rg1f  = (float*)alloc((size_t)B_ * S_ * (H_ / 2) * 4);
  float* wg1f  = (float*)alloc((size_t)B_ * S_ * (H_ / 2) * 4);
  float* c2f   = (float*)alloc((size_t)B_ * L_ * H_ * 4);
  float* wgv   = (float*)alloc((size_t)B_ * S_ * 4);

  dim3 blk256(256);
  // ---- converts / weight transposes ----
  cvt_f32_bf16_kernel<<<(B_*S_*H_)/256, blk256, 0, stream>>>(hs,   hsb,   B_*S_*H_);
  cvt_f32_bf16_kernel<<<(B_*M_*H_)/256, blk256, 0, stream>>>(bank, bankb, B_*M_*H_);
  wtrans_bf16_kernel<<<(H_*H_)/256,        blk256, 0, stream>>>(Wq,   WqT,   H_, H_);
  wtrans_bf16_kernel<<<(H_*H_)/256,        blk256, 0, stream>>>(Wk,   WkT,   H_, H_);
  wtrans_bf16_kernel<<<(H_*H_)/256,        blk256, 0, stream>>>(Wv,   WvT,   H_, H_);
  wtrans_bf16_kernel<<<(H_*H_)/256,        blk256, 0, stream>>>(Wo,   WoT,   H_, H_);
  wtrans_bf16_kernel<<<(2*H_*(H_/2))/256,  blk256, 0, stream>>>(Wrg1, Wrg1T, 2*H_, H_/2);
  wtrans_bf16_kernel<<<(H_*(H_/2))/256,    blk256, 0, stream>>>(Wwg1, Wwg1T, H_,   H_/2);
  wtrans_bf16_kernel<<<(CR_*H_*H_)/256,    blk256, 0, stream>>>(Wc1,  Wc1T,  CR_*H_, H_);
  wtrans_bf16_kernel<<<(H_*H_)/256,        blk256, 0, stream>>>(Wc2,  Wc2T,  H_, H_);

  auto gemm = [&](const bf16* A0, int ld0, int Ksplit, const bf16* A1, int ld1,
                  const bf16* Wt, const float* bias, float* C32, bf16* Cbf,
                  int M, int N, int K, int act) {
    dim3 grid(N / 128, M / 128);
    gemm_wmma_bf16_kernel<<<grid, blk256, 0, stream>>>(A0, ld0, Ksplit, A1, ld1,
                                                       Wt, bias, C32, Cbf, M, N, K, act);
  };

  // ---- projections ----
  gemm(hsb,   H_, H_, nullptr, 0, WqT, bq, nullptr, qb,  B_*S_, H_, H_, 0);
  gemm(bankb, H_, H_, nullptr, 0, WkT, bk, nullptr, kb_, B_*M_, H_, H_, 0);
  gemm(bankb, H_, H_, nullptr, 0, WvT, bv, nullptr, vb_, B_*M_, H_, H_, 0);

  // ---- attention ----
  attn_wmma_kernel<<<(B_*NH_*(S_/16))/4, dim3(128), 0, stream>>>(qb, kb_, vb_, mask, attb);

  // ---- output projection (f32 for LN path, bf16 for gate concat) ----
  gemm(attb, H_, H_, nullptr, 0, WoT, bo, retf, retb, B_*S_, H_, H_, 0);

  // ---- gates + compressor ----
  gemm(hsb, H_, H_, retb, H_, Wrg1T, brg1, rg1f, nullptr, B_*S_, H_/2, 2*H_, 1);
  gemm(hsb, H_, H_, nullptr, 0, Wwg1T, bwg1, wg1f, nullptr, B_*S_, H_/2, H_, 1);
  gemm(hsb, CR_*H_, CR_*H_, nullptr, 0, Wc1T, bc1, nullptr, c1b, B_*L_, H_, CR_*H_, 1);
  gemm(c1b, H_, H_, nullptr, 0, Wc2T, bc2, c2f, nullptr, B_*L_, H_, H_, 0);

  // ---- fused read-gate + residual + LayerNorm -> out ----
  rg_ln_kernel<<<B_*S_, blk256, 0, stream>>>(hs, retf, rg1f, Wrg2, brg2, lng, lnb, out_main);

  // ---- write gate + memory update ----
  wg_kernel<<<(B_*S_)/8, blk256, 0, stream>>>(wg1f, Wwg2, bwg2, wgv);
  (void)hipMemcpyAsync(out_bank, bank, (size_t)B_*M_*H_*sizeof(float),
                       hipMemcpyDeviceToDevice, stream);
  scatter_kernel<<<B_*L_, blk256, 0, stream>>>(c2f, wgv, wp, out_bank);
  maskptr_kernel<<<(B_*M_ + 255)/256, blk256, 0, stream>>>(mask, wp, out_mask, out_ptr);
}
